// Morpher_1898375545537
// MI455X (gfx1250) — compile-verified, matrix operand order checked by reading, not run on hardware
//
#include <hip/hip_runtime.h>
#include <hip/hip_bf16.h>

typedef __bf16 bf16_t;
typedef __bf16 v16bf __attribute__((ext_vector_type(16)));
typedef float  v8f  __attribute__((ext_vector_type(8)));

#define KTS    3
#define NSTR   4
#define DHEAD  64
#define DMODEL 448
#define IODIM  512
#define RRANK  64
#define HIDDIM 1792
#define BB     8
#define TT     1024
#define BTOK   (BB*TT)      /* 8192 */
#define HTOT   12           /* N*K heads */
#define EPSF   1e-6f

// ---------------- helpers ----------------

__device__ __forceinline__ unsigned bfbits(float x){   // bf16 (RNE) in low 16 bits
  unsigned u = __float_as_uint(x);
  return (u + 0x7FFFu + ((u >> 16) & 1u)) >> 16;
}
__device__ __forceinline__ unsigned packbf2(float lo, float hi){
  return bfbits(lo) | (bfbits(hi) << 16);
}
__device__ __forceinline__ bf16_t f2bf(float x){
  unsigned short s = (unsigned short)bfbits(x);
  return __builtin_bit_cast(bf16_t, s);
}
__device__ __forceinline__ v8f zero8(){
  v8f a;
  #pragma unroll
  for (int i = 0; i < 8; ++i) a[i] = 0.f;
  return a;
}

__device__ __forceinline__ void make_tables(int ph, int* slots, int* perm, int* inv){
  const int ts[KTS] = {1, 2, 4};
  int off = 0;
  #pragma unroll
  for (int k = 0; k < KTS; ++k){
    int j = ph % ts[k];
    slots[k] = off + j;
    off += ts[k];
    int alpha = NSTR / ts[k];
    #pragma unroll
    for (int n = 0; n < NSTR; ++n) perm[k*NSTR + n] = (n + alpha*j) & (NSTR - 1);
    #pragma unroll
    for (int n = 0; n < NSTR; ++n) inv[k*NSTR + perm[k*NSTR + n]] = n;
  }
}

// ---------------- generic batched bf16-WMMA GEMM ----------------
// Requirements (all call sites satisfy): M % 128 == 0, Kd % 64 == 0,
// N % 4 == 0, lda/ldb % 4 == 0.
// mode 0:  C = alpha*acc + Add   (Add optional)
// mode 1:  C = silu(Add) * acc   (gated-MLP fusion, in-place safe)

struct GemmP {
  const float* A;   long lda,  sA;
  const float* Bm;  long ldb,  sB;
  float*       C;   long ldc,  sC;
  const float* Add; long ldadd, sAdd;
  long M, N, Kd;
  float alpha;
  const float* alphaPtr;
  int mode;
};

__launch_bounds__(256)
__global__ void gemm_bf16_wmma(GemmP p){
  __shared__ __attribute__((aligned(64))) bf16_t sA[128*64];   // [row][k]
  __shared__ __attribute__((aligned(64))) bf16_t sB[128*64];   // [col][k] (transposed)

  const int tid  = threadIdx.x;
  const int lane = tid & 31;
  const int w    = tid >> 5;
  const int wm   = w & 1;
  const int wn   = w >> 1;
  const long batch = blockIdx.z;
  const long row0  = (long)blockIdx.y * 128;
  const long col0  = (long)blockIdx.x * 128;

  const float* Ab = p.A  + batch * p.sA;
  const float* Bb = p.Bm + batch * p.sB;

  // staging roles
  const int arow = tid >> 1, acb = (tid & 1) * 32;   // A: 1 row, 32 cols
  const int bc4  = (tid & 31) * 4;                   // B: 4 cols, 8 k
  const int bkb  = (tid >> 5) * 8;

  v8f acc[4][2];
  #pragma unroll
  for (int mf = 0; mf < 4; ++mf)
    #pragma unroll
    for (int nf = 0; nf < 2; ++nf) acc[mf][nf] = zero8();

  const long gcB  = col0 + bc4;
  const long gcBc = (gcB < p.N) ? gcB : (p.N - 4);
  const bool bOOB = (gcB >= p.N);

  for (long k0 = 0; k0 < p.Kd; k0 += 64){
    if (k0 + 64 < p.Kd){   // speculative prefetch of next K tile (global_prefetch_b8)
      __builtin_prefetch(&Ab[(row0 + arow) * p.lda + k0 + 64 + acb], 0, 1);
      __builtin_prefetch(&Bb[(k0 + 64 + bkb) * p.ldb + gcBc], 0, 1);
    }
    // ---- stage A (128x64): 8 x b128 loads -> 4 x b128 LDS stores ----
    {
      const float* arp = Ab + (row0 + arow) * p.lda + k0 + acb;
      #pragma unroll
      for (int i = 0; i < 4; ++i){
        float4 v0 = ((const float4*)arp)[2*i];
        float4 v1 = ((const float4*)arp)[2*i + 1];
        uint4 o;
        o.x = packbf2(v0.x, v0.y); o.y = packbf2(v0.z, v0.w);
        o.z = packbf2(v1.x, v1.y); o.w = packbf2(v1.z, v1.w);
        *(uint4*)&sA[arow*64 + acb + i*8] = o;
      }
    }
    // ---- stage B (64x128) transposed: 8 x b128 loads -> 4 x b128 LDS stores ----
    {
      float rr[8][4];
      #pragma unroll
      for (int kk = 0; kk < 8; ++kk){
        float4 v = *(const float4*)&Bb[(k0 + bkb + kk) * p.ldb + gcBc];
        rr[kk][0] = v.x; rr[kk][1] = v.y; rr[kk][2] = v.z; rr[kk][3] = v.w;
      }
      if (bOOB){
        #pragma unroll
        for (int kk = 0; kk < 8; ++kk)
          #pragma unroll
          for (int c = 0; c < 4; ++c) rr[kk][c] = 0.f;
      }
      #pragma unroll
      for (int c = 0; c < 4; ++c){
        uint4 o;
        o.x = packbf2(rr[0][c], rr[1][c]);
        o.y = packbf2(rr[2][c], rr[3][c]);
        o.z = packbf2(rr[4][c], rr[5][c]);
        o.w = packbf2(rr[6][c], rr[7][c]);
        *(uint4*)&sB[(bc4 + c)*64 + bkb] = o;
      }
    }
    __syncthreads();

    v16bf afr[4][2], bfr[2][2];
    #pragma unroll
    for (int mf = 0; mf < 4; ++mf)
      #pragma unroll
      for (int kf = 0; kf < 2; ++kf)
        afr[mf][kf] = *(const v16bf*)&sA[(wm*64 + mf*16 + (lane & 15))*64 + kf*32 + (lane >> 4)*16];
    #pragma unroll
    for (int nf = 0; nf < 2; ++nf)
      #pragma unroll
      for (int kf = 0; kf < 2; ++kf)
        bfr[nf][kf] = *(const v16bf*)&sB[(wn*32 + nf*16 + (lane & 15))*64 + kf*32 + (lane >> 4)*16];

    #pragma unroll
    for (int kf = 0; kf < 2; ++kf)
      #pragma unroll
      for (int mf = 0; mf < 4; ++mf)
        #pragma unroll
        for (int nf = 0; nf < 2; ++nf)
          acc[mf][nf] = __builtin_amdgcn_wmma_f32_16x16x32_bf16(
              false, afr[mf][kf], false, bfr[nf][kf], (short)0, acc[mf][nf], false, false);
    __syncthreads();
  }

  float alpha = p.alpha;
  if (p.alphaPtr) alpha *= *p.alphaPtr;
  float*       Cb  = p.C + batch * p.sC;
  const float* Adb = p.Add ? (p.Add + batch * p.sAdd) : nullptr;

  #pragma unroll
  for (int mf = 0; mf < 4; ++mf){
    #pragma unroll
    for (int nf = 0; nf < 2; ++nf){
      #pragma unroll
      for (int r = 0; r < 8; ++r){
        long row = row0 + wm*64 + mf*16 + ((lane >> 4) << 3) + r;
        long col = col0 + wn*32 + nf*16 + (lane & 15);
        if (col < p.N){
          float v = acc[mf][nf][r];
          if (p.mode == 1){
            float a = Adb[row * p.ldadd + col];
            float s = a / (1.f + __expf(-a));
            Cb[row * p.ldc + col] = s * v;
          } else {
            float o = alpha * v;
            if (Adb) o += Adb[row * p.ldadd + col];
            Cb[row * p.ldc + col] = o;
          }
        }
      }
    }
  }
}

// ---------------- RMS + slot gather: x_head[k][n][bt][d] ----------------

__global__ void rms_gather_xhead(const float* __restrict__ z, float* __restrict__ xh,
                                 const int* __restrict__ phase){
  int slots[KTS], perm[KTS*NSTR], inv[KTS*NSTR];
  make_tables(*phase, slots, perm, inv);
  (void)perm;
  long bid = blockIdx.x;                 // bt*12 + k*4 + n
  int n = bid & 3; int k = (int)((bid >> 2) % 3); long bt = bid / 12;
  const float* src = z + bt * (long)(NSTR*DMODEL) + (long)inv[k*NSTR + n] * DMODEL
                       + slots[k] * DHEAD;
  int lane = threadIdx.x;
  float a0 = src[lane], a1 = src[lane + 32];
  float ss = a0*a0 + a1*a1;
  #pragma unroll
  for (int m = 16; m >= 1; m >>= 1) ss += __shfl_xor(ss, m);
  float sc = rsqrtf(ss * (1.f/64.f) + EPSF);
  float* dst = xh + ((long)(k*NSTR + n) * BTOK + bt) * DHEAD;
  dst[lane] = a0 * sc; dst[lane + 32] = a1 * sc;
}

// ---------------- qkv split + head permute + bf16 pack (4-wide) ----------------

__global__ void pack_qkv(const float* __restrict__ qkv, bf16_t* __restrict__ qp,
                         bf16_t* __restrict__ kp, bf16_t* __restrict__ vp,
                         const int* __restrict__ phase){
  int slots[KTS], perm[KTS*NSTR], inv[KTS*NSTR];
  make_tables(*phase, slots, perm, inv);
  (void)slots; (void)inv;
  long g = (long)blockIdx.x * blockDim.x + threadIdx.x;   // over total/4
  if (g >= (long)BB*HTOT*TT*(DHEAD/4)) return;
  int dq  = (int)(g & 15);            // which float4 within the 64-d head
  long t2 = g >> 4;                   // b*H*T + h*T + t
  int t  = (int)(t2 % TT);  long t3 = t2 / TT;
  int h  = (int)(t3 % HTOT); int b = (int)(t3 / HTOT);
  int n = h / KTS, k = h % KTS;
  int np = perm[k*NSTR + n];
  long bt = (long)b*TT + t;
  const float* src = qkv + ((long)(k*NSTR + np) * BTOK + bt) * (3*DHEAD);
  float4 q4 = ((const float4*)src)[dq];
  float4 k4 = ((const float4*)(src + DHEAD))[dq];
  float4 v4 = ((const float4*)(src + 2*DHEAD))[dq];
  long o2 = t2*16 + dq;               // uint2 index
  ((uint2*)qp)[o2] = make_uint2(packbf2(q4.x*0.125f, q4.y*0.125f), packbf2(q4.z*0.125f, q4.w*0.125f));
  ((uint2*)kp)[o2] = make_uint2(packbf2(k4.x, k4.y), packbf2(k4.z, k4.w));
  ((uint2*)vp)[o2] = make_uint2(packbf2(v4.x, v4.y), packbf2(v4.z, v4.w));
}

// ---------------- causal flash attention (WMMA bf16) ----------------

__launch_bounds__(128)
__global__ void flash_attn(const bf16_t* __restrict__ qp, const bf16_t* __restrict__ kp,
                           const bf16_t* __restrict__ vp, float* __restrict__ attn){
  __shared__ __attribute__((aligned(64))) bf16_t sK [64*64];   // [key][dv]
  __shared__ __attribute__((aligned(64))) bf16_t sVt[64*64];   // [dv][key]
  __shared__ __attribute__((aligned(64))) bf16_t sP [64*64];   // [qrow][key]

  const int tid = threadIdx.x, lane = tid & 31, wq = tid >> 5;
  const int qt = blockIdx.x;
  const long hb = ((long)blockIdx.z * HTOT + blockIdx.y) * TT;
  const int q0 = qt * 64;

  v16bf aq[2];
  {
    const bf16_t* qrow = qp + (hb + q0 + wq*16 + (lane & 15)) * DHEAD + (lane >> 4)*16;
    aq[0] = *(const v16bf*)(qrow);
    aq[1] = *(const v16bf*)(qrow + 32);
  }

  v8f o[4];
  #pragma unroll
  for (int df = 0; df < 4; ++df) o[df] = zero8();
  float mrow[8], lrow[8];
  #pragma unroll
  for (int r = 0; r < 8; ++r){ mrow[r] = -1e30f; lrow[r] = 0.f; }

  const int kkey = tid >> 1, kdvb = (tid & 1) * 32;   // K staging role
  const int vkb  = (tid & 15) * 4, vdvb = (tid >> 4) * 8; // V staging role

  for (int kt = 0; kt <= qt; ++kt){
    // ---- stage K tile row-major (uint4 copies) ----
    {
      const uint4* src = (const uint4*)(kp + (hb + (long)kt*64 + kkey)*DHEAD + kdvb);
      uint4* dst = (uint4*)&sK[kkey*64 + kdvb];
      #pragma unroll
      for (int i = 0; i < 4; ++i) dst[i] = src[i];
    }
    // ---- stage V tile transposed: 4 keys x 8 dv register block ----
    {
      uint4 rv[4];
      #pragma unroll
      for (int kk = 0; kk < 4; ++kk)
        rv[kk] = *(const uint4*)(vp + (hb + (long)kt*64 + vkb + kk)*DHEAD + vdvb);
      const unsigned* w0 = (const unsigned*)&rv[0];
      const unsigned* w1 = (const unsigned*)&rv[1];
      const unsigned* w2 = (const unsigned*)&rv[2];
      const unsigned* w3 = (const unsigned*)&rv[3];
      #pragma unroll
      for (int i = 0; i < 8; ++i){
        int wi = i >> 1, sh = (i & 1) * 16;
        unsigned h0 = (w0[wi] >> sh) & 0xFFFFu;
        unsigned h1 = (w1[wi] >> sh) & 0xFFFFu;
        unsigned h2 = (w2[wi] >> sh) & 0xFFFFu;
        unsigned h3 = (w3[wi] >> sh) & 0xFFFFu;
        *(uint2*)&sVt[(vdvb + i)*64 + vkb] = make_uint2(h0 | (h1 << 16), h2 | (h3 << 16));
      }
    }
    __syncthreads();

    v8f s[4];
    #pragma unroll
    for (int nf = 0; nf < 4; ++nf){
      s[nf] = zero8();
      #pragma unroll
      for (int kf = 0; kf < 2; ++kf){
        v16bf bk = *(const v16bf*)&sK[(nf*16 + (lane & 15))*64 + kf*32 + (lane >> 4)*16];
        s[nf] = __builtin_amdgcn_wmma_f32_16x16x32_bf16(
            false, aq[kf], false, bk, (short)0, s[nf], false, false);
      }
    }

    if (kt == qt){  // causal mask on diagonal tile
      #pragma unroll
      for (int nf = 0; nf < 4; ++nf)
        #pragma unroll
        for (int r = 0; r < 8; ++r){
          int gr = q0 + wq*16 + ((lane >> 4) << 3) + r;
          int gc = kt*64 + nf*16 + (lane & 15);
          if (gc > gr) s[nf][r] = -1e30f;
        }
    }

    #pragma unroll
    for (int r = 0; r < 8; ++r){
      float rm = fmaxf(fmaxf(s[0][r], s[1][r]), fmaxf(s[2][r], s[3][r]));
      #pragma unroll
      for (int msk = 8; msk >= 1; msk >>= 1) rm = fmaxf(rm, __shfl_xor(rm, msk));
      float mn  = fmaxf(mrow[r], rm);
      float fac = __expf(mrow[r] - mn);
      float rs = 0.f;
      #pragma unroll
      for (int nf = 0; nf < 4; ++nf){
        float pv = __expf(s[nf][r] - mn);
        s[nf][r] = pv;
        rs += pv;
      }
      #pragma unroll
      for (int msk = 8; msk >= 1; msk >>= 1) rs += __shfl_xor(rs, msk);
      lrow[r] = lrow[r]*fac + rs;
      mrow[r] = mn;
      #pragma unroll
      for (int df = 0; df < 4; ++df) o[df][r] *= fac;
      int prow = wq*16 + ((lane >> 4) << 3) + r;
      #pragma unroll
      for (int nf = 0; nf < 4; ++nf)
        sP[prow*64 + nf*16 + (lane & 15)] = f2bf(s[nf][r]);
    }
    __syncthreads();

    #pragma unroll
    for (int kf = 0; kf < 2; ++kf){
      v16bf ap = *(const v16bf*)&sP[(wq*16 + (lane & 15))*64 + kf*32 + (lane >> 4)*16];
      #pragma unroll
      for (int df = 0; df < 4; ++df){
        v16bf bv = *(const v16bf*)&sVt[(df*16 + (lane & 15))*64 + kf*32 + (lane >> 4)*16];
        o[df] = __builtin_amdgcn_wmma_f32_16x16x32_bf16(
            false, ap, false, bv, (short)0, o[df], false, false);
      }
    }
    __syncthreads();
  }

  #pragma unroll
  for (int r = 0; r < 8; ++r){
    int gr = q0 + wq*16 + ((lane >> 4) << 3) + r;
    float invl = 1.f / lrow[r];
    #pragma unroll
    for (int df = 0; df < 4; ++df)
      attn[(hb + gr) * DHEAD + df*16 + (lane & 15)] = o[df][r] * invl;
  }
}

// ---------------- scatter attention back into z slots (4-wide) ----------------

__global__ void scatter_attn(const float* __restrict__ attn, float* __restrict__ z,
                             const int* __restrict__ phase){
  int slots[KTS], perm[KTS*NSTR], inv[KTS*NSTR];
  make_tables(*phase, slots, perm, inv);
  (void)perm; (void)inv;
  long g = (long)blockIdx.x * blockDim.x + threadIdx.x;   // over total/4
  if (g >= (long)BB*HTOT*TT*(DHEAD/4)) return;
  int dq  = (int)(g & 15);
  long t2 = g >> 4;
  int t  = (int)(t2 % TT);  long t3 = t2 / TT;
  int h  = (int)(t3 % HTOT); int b = (int)(t3 / HTOT);
  int n = h / KTS, k = h % KTS;
  long bt = (long)b*TT + t;
  float4 a = ((const float4*)attn)[t2*16 + dq];
  float4* zp = (float4*)&z[bt*(long)(NSTR*DMODEL) + (long)n*DMODEL + slots[k]*DHEAD + dq*4];
  float4 c = *zp;
  c.x += a.x; c.y += a.y; c.z += a.z; c.w += a.w;
  *zp = c;
}

// ---------------- wave-level row RMS norm ----------------

__global__ void rms_rows(const float* __restrict__ src, float* __restrict__ dst, int len){
  long row = blockIdx.x;
  const float* s = src + row * (long)len;
  float ss = 0.f;
  for (int i = threadIdx.x; i < len; i += 32){ float v = s[i]; ss += v*v; }
  #pragma unroll
  for (int m = 16; m >= 1; m >>= 1) ss += __shfl_xor(ss, m);
  float sc = rsqrtf(ss / (float)len + EPSF);
  float* d = dst + row * (long)len;
  for (int i = threadIdx.x; i < len; i += 32) d[i] = s[i] * sc;
}

// ---------------- host orchestration ----------------

static void launch_gemm(const GemmP& p, int batch, hipStream_t st){
  dim3 g((unsigned)((p.N + 127) / 128), (unsigned)((p.M + 127) / 128), (unsigned)batch);
  gemm_bf16_wmma<<<g, 256, 0, st>>>(p);
}

extern "C" void kernel_launch(void* const* d_in, const int* in_sizes, int n_in,
                              void* d_out, int out_size, void* d_ws, size_t ws_size,
                              hipStream_t stream){
  (void)in_sizes; (void)n_in; (void)out_size; (void)ws_size;

  const float* x     = (const float*)d_in[0];
  const float* Wbase = (const float*)d_in[1];
  const float* WA    = (const float*)d_in[2];
  const float* encB  = (const float*)d_in[3];
  const float* beta  = (const float*)d_in[4];
  const float* Wqkv  = (const float*)d_in[5];
  const float* Wg    = (const float*)d_in[6];
  const float* Wu    = (const float*)d_in[7];
  const float* Wd    = (const float*)d_in[8];
  const float* Wdown = (const float*)d_in[9];
  const float* Wup   = (const float*)d_in[10];
  const int*   phase = (const int*)d_in[11];
  float* out = (float*)d_out;

  float* ws = (float*)d_ws;
  size_t off = 0;
  auto alloc = [&](size_t nfloats){ float* p = ws + off; off += nfloats; return p; };

  float*  f_h    = alloc((size_t)BTOK * RRANK);
  float*  f_base = alloc((size_t)BTOK * DMODEL);
  float*  f_z    = alloc((size_t)BTOK * NSTR * DMODEL);
  float*  f_xh   = alloc((size_t)HTOT * BTOK * DHEAD);
  float*  f_qkv  = alloc((size_t)HTOT * BTOK * 3 * DHEAD);
  bf16_t* f_qp   = (bf16_t*)alloc((size_t)BB*HTOT*TT*DHEAD/2);
  bf16_t* f_kp   = (bf16_t*)alloc((size_t)BB*HTOT*TT*DHEAD/2);
  bf16_t* f_vp   = (bf16_t*)alloc((size_t)BB*HTOT*TT*DHEAD/2);
  float*  f_attn = alloc((size_t)BB*HTOT*TT*DHEAD);
  float*  f_hm   = alloc((size_t)BTOK * NSTR * DMODEL);
  float*  f_g    = alloc((size_t)BTOK * NSTR * HIDDIM);
  float*  f_cat  = f_qkv;
  float*  f_r    = f_h;

  // 1) h = x @ WA
  { GemmP p{ x, IODIM, 0, WA, RRANK, 0, f_h, RRANK, 0, nullptr, 0, 0,
             BTOK, RRANK, IODIM, 1.f, nullptr, 0 }; launch_gemm(p, 1, stream); }
  // 2) base = x @ Wbase
  { GemmP p{ x, IODIM, 0, Wbase, DMODEL, 0, f_base, DMODEL, 0, nullptr, 0, 0,
             BTOK, DMODEL, IODIM, 1.f, nullptr, 0 }; launch_gemm(p, 1, stream); }
  // 3) z[:, n, :] = base + beta * (h @ encB[n])
  { GemmP p{ f_h, RRANK, 0, encB, DMODEL, (long)RRANK*DMODEL,
             f_z, (long)NSTR*DMODEL, DMODEL, f_base, DMODEL, 0,
             BTOK, DMODEL, RRANK, 1.f, beta, 0 }; launch_gemm(p, NSTR, stream); }
  // 4) RMS + slot gather
  rms_gather_xhead<<<BTOK * HTOT, 32, 0, stream>>>(f_z, f_xh, phase);
  // 5) qkv[k][n] = x_head[k][n] @ Wqkv[k][n]
  { GemmP p{ f_xh, DHEAD, (long)BTOK*DHEAD, Wqkv, 3*DHEAD, (long)DHEAD*3*DHEAD,
             f_qkv, 3*DHEAD, (long)BTOK*3*DHEAD, nullptr, 0, 0,
             BTOK, 3*DHEAD, DHEAD, 1.f, nullptr, 0 }; launch_gemm(p, HTOT, stream); }
  // 6) split + permute + pack bf16
  { long tot = (long)BB*HTOT*TT*(DHEAD/4);
    pack_qkv<<<(unsigned)((tot + 255) / 256), 256, 0, stream>>>(f_qkv, f_qp, f_kp, f_vp, phase); }
  // 7) causal flash attention
  flash_attn<<<dim3(TT/64, HTOT, BB), 128, 0, stream>>>(f_qp, f_kp, f_vp, f_attn);
  // 8) scatter attention into z slots
  { long tot = (long)BB*HTOT*TT*(DHEAD/4);
    scatter_attn<<<(unsigned)((tot + 255) / 256), 256, 0, stream>>>(f_attn, f_z, phase); }
  // 9) hm = rms(z) per (bt, n)
  rms_rows<<<BTOK * NSTR, 32, 0, stream>>>(f_z, f_hm, DMODEL);
  // 10) g = hm @ Wg
  { GemmP p{ f_hm, DMODEL, 0, Wg, HIDDIM, 0, f_g, HIDDIM, 0, nullptr, 0, 0,
             (long)BTOK*NSTR, HIDDIM, DMODEL, 1.f, nullptr, 0 }; launch_gemm(p, 1, stream); }
  // 11) g = silu(g) * (hm @ Wu)   (fused, in place)
  { GemmP p{ f_hm, DMODEL, 0, Wu, HIDDIM, 0, f_g, HIDDIM, 0, f_g, HIDDIM, 0,
             (long)BTOK*NSTR, HIDDIM, DMODEL, 1.f, nullptr, 1 }; launch_gemm(p, 1, stream); }
  // 12) z += g @ Wd   (in-place residual)
  { GemmP p{ f_g, HIDDIM, 0, Wd, DMODEL, 0, f_z, DMODEL, 0, f_z, DMODEL, 0,
             (long)BTOK*NSTR, DMODEL, HIDDIM, 1.f, nullptr, 0 }; launch_gemm(p, 1, stream); }
  // 13) cat = rms(z) per bt over N*D
  rms_rows<<<BTOK, 32, 0, stream>>>(f_z, f_cat, NSTR * DMODEL);
  // 14) r = cat @ Wdown
  { GemmP p{ f_cat, (long)NSTR*DMODEL, 0, Wdown, RRANK, 0, f_r, RRANK, 0, nullptr, 0, 0,
             BTOK, RRANK, (long)NSTR*DMODEL, 1.f, nullptr, 0 }; launch_gemm(p, 1, stream); }
  // 15) out = r @ Wup
  { GemmP p{ f_r, RRANK, 0, Wup, IODIM, 0, out, IODIM, 0, nullptr, 0, 0,
             BTOK, IODIM, RRANK, 1.f, nullptr, 0 }; launch_gemm(p, 1, stream); }
}
